// FAVORAttention_23837068493288
// MI455X (gfx1250) — matrix operand order repaired and checked by reading, not compiled, not run on hardware
//
#include <hip/hip_runtime.h>

// Problem constants (match reference)
#define BB   4
#define NN   4096
#define DD   1024
#define HH   16
#define MM   256
#define DKK  64
#define TD   (3 * DD)     // 3072
#define BN   (BB * NN)    // 16384
#define EPSF 1e-6f

typedef __attribute__((ext_vector_type(16))) __bf16 v16bf;
typedef __attribute__((ext_vector_type(8)))  __bf16 v8bf;
typedef __attribute__((ext_vector_type(8)))  float  v8f;

__device__ __forceinline__ v8f wmma_bf16(v16bf a, v16bf b, v8f c) {
  return __builtin_amdgcn_wmma_f32_16x16x32_bf16(
      false, a, false, b, (short)0, c, false, false);
}

// ---- Fragment loaders (ISA 7.12.2 layouts, wave32), 16B vector loads ------
// A 16x32 bf16 from f32 row-major (only used where A is produced in f32).
__device__ __forceinline__ v16bf load_a_f32(const float* __restrict__ Abase,
                                            int ld, int k0, int lane) {
  const float* row = Abase + (size_t)(lane & 15) * (size_t)ld;
  int half = (lane >> 4) & 1;
  v16bf f;
#pragma unroll
  for (int g = 0; g < 2; ++g) {
    const float4* p = (const float4*)(row + k0 + g * 16 + half * 8);
    float4 u = p[0], w = p[1];
    f[8*g+0] = (__bf16)u.x; f[8*g+1] = (__bf16)u.y;
    f[8*g+2] = (__bf16)u.z; f[8*g+3] = (__bf16)u.w;
    f[8*g+4] = (__bf16)w.x; f[8*g+5] = (__bf16)w.y;
    f[8*g+6] = (__bf16)w.z; f[8*g+7] = (__bf16)w.w;
  }
  return f;
}

// A 16x32 bf16 from bf16 row-major: two 16B loads, zero converts.
__device__ __forceinline__ v16bf load_a_bf16(const __bf16* __restrict__ Abase,
                                             int ld, int k0, int lane) {
  const __bf16* row = Abase + (size_t)(lane & 15) * (size_t)ld;
  int half = (lane >> 4) & 1;
  v8bf lo = *(const v8bf*)(row + k0 + half * 8);
  v8bf hi = *(const v8bf*)(row + k0 + 16 + half * 8);
  return __builtin_shufflevector(lo, hi, 0,1,2,3,4,5,6,7,8,9,10,11,12,13,14,15);
}

// B 32x16 bf16 where B[k][j] = src[j][k] (src bf16 row-major, ld = row stride):
// frag elem i <-> src row j, 8 bf16 at k0 + 8*(lane>>4) then 8 at k0 + 16 + 8*(lane>>4).
__device__ __forceinline__ v16bf load_b_bf16t(const __bf16* __restrict__ Bbase,
                                              int ld, int k0, int lane) {
  const __bf16* row = Bbase + (size_t)(lane & 15) * (size_t)ld;
  int half = (lane >> 4) & 1;
  v8bf lo = *(const v8bf*)(row + k0 + half * 8);
  v8bf hi = *(const v8bf*)(row + k0 + 16 + half * 8);
  return __builtin_shufflevector(lo, hi, 0,1,2,3,4,5,6,7,8,9,10,11,12,13,14,15);
}

// ---- Kernel 0: elementwise f32 -> bf16 (8 elems / thread) -----------------
__global__ void cvt_kernel(const float* __restrict__ src, __bf16* __restrict__ dst,
                           int n8) {
  int tid = blockIdx.x * blockDim.x + threadIdx.x;
  if (tid >= n8) return;
  const float4* p = (const float4*)(src + (size_t)tid * 8);
  float4 u = p[0], w = p[1];
  v8bf t;
  t[0]=(__bf16)u.x; t[1]=(__bf16)u.y; t[2]=(__bf16)u.z; t[3]=(__bf16)u.w;
  t[4]=(__bf16)w.x; t[5]=(__bf16)w.y; t[6]=(__bf16)w.z; t[7]=(__bf16)w.w;
  *(v8bf*)(dst + (size_t)tid * 8) = t;
}

// ---- Kernel 1: qkv = x @ Wqkv^T + bqkv, mask Q/V sections -----------------
// One wave -> 16 rows x 64 cols (4 col tiles share the A fragment).
__global__ void qkv_kernel(const __bf16* __restrict__ xb, const __bf16* __restrict__ Wb,
                           const float* __restrict__ bias, const int* __restrict__ mask,
                           float* __restrict__ qkv) {
  int wid  = (int)((blockIdx.x * blockDim.x + threadIdx.x) >> 5);
  int lane = threadIdx.x & 31;
  const int RT = BN / 16, CTG = TD / 64;  // 1024 x 48
  if (wid >= RT * CTG) return;
  int rt = wid % RT, ct0 = (wid / RT) * 4;

  const __bf16* Abase = xb + (size_t)rt * 16 * DD;
  v8f acc[4] = {};
  for (int k0 = 0; k0 < DD; k0 += 32) {
    v16bf a = load_a_bf16(Abase, DD, k0, lane);
    __builtin_prefetch(Abase + (size_t)(lane & 15) * DD + k0 + 128, 0, 1);
#pragma unroll
    for (int cc = 0; cc < 4; ++cc)
      acc[cc] = wmma_bf16(a, load_b_bf16t(Wb + (size_t)(ct0 + cc) * 16 * DD, DD, k0, lane), acc[cc]);
  }
  int c = lane & 15, half = lane >> 4;
  float bj[4]; bool mb[4];
#pragma unroll
  for (int cc = 0; cc < 4; ++cc) {
    int j = (ct0 + cc) * 16 + c;
    bj[cc] = bias[j];
    mb[cc] = (j < DD) || (j >= 2 * DD);   // Q or V section
  }
#pragma unroll
  for (int v = 0; v < 8; ++v) {
    int rowg = rt * 16 + v + 8 * half;           // flat (b,n) == mask index
    float keep = mask[rowg] ? 0.0f : 1.0f;
#pragma unroll
    for (int cc = 0; cc < 4; ++cc) {
      float val = acc[cc][v] + bj[cc];
      if (mb[cc]) val *= keep;
      qkv[(size_t)rowg * TD + (ct0 + cc) * 16 + c] = val;
    }
  }
}

// ---- Kernel 1b: Vt[b,h,d,n] = bf16(V) via LDS-tiled transpose -------------
__global__ void vt_kernel(const float* __restrict__ qkv, __bf16* __restrict__ Vt) {
  const int NB = NN / 32;                 // 128 n-blocks
  int nblk = blockIdx.x % NB;
  int bh   = blockIdx.x / NB;
  int h = bh % HH, b = bh / HH;
  __shared__ __bf16 sm[64 * 32];          // [d][n_local]
  int i = threadIdx.x;
  {
    int nl = i >> 3, part = i & 7;        // 32 rows x 8 chunks of 8 d's
    const float* src = qkv + ((size_t)b * NN + (size_t)nblk * 32 + nl) * TD
                       + 2 * DD + h * DKK + part * 8;
    float4 u = *(const float4*)(src);
    float4 w = *(const float4*)(src + 4);
    int d0 = part * 8;
    sm[(d0+0)*32 + nl] = (__bf16)u.x; sm[(d0+1)*32 + nl] = (__bf16)u.y;
    sm[(d0+2)*32 + nl] = (__bf16)u.z; sm[(d0+3)*32 + nl] = (__bf16)u.w;
    sm[(d0+4)*32 + nl] = (__bf16)w.x; sm[(d0+5)*32 + nl] = (__bf16)w.y;
    sm[(d0+6)*32 + nl] = (__bf16)w.z; sm[(d0+7)*32 + nl] = (__bf16)w.w;
  }
  __syncthreads();
  {
    int d = i >> 2, q = i & 3;            // 64 d-rows x 4 chunks of 8 n's
    v8bf t;
    const __bf16* rp = sm + d * 32 + q * 8;
#pragma unroll
    for (int jj = 0; jj < 8; ++jj) t[jj] = rp[jj];
    *(v8bf*)(Vt + ((size_t)bh * DKK + d) * NN + (size_t)nblk * 32 + q * 8) = t;
  }
}

// ---- Kernel 2: phi = exp(proj - rowmax - 0.5|t|^2)/sqrt(M) ----------------
// isK==0: Qf row-major (b,h,n,m). isK==1: Kft (b,h,m,n), masked.
// LDS-staged coalesced stores. 128 threads (4 waves) per block.
__global__ void __launch_bounds__(128) phi_kernel(
    const float* __restrict__ qkv, const __bf16* __restrict__ omegab,
    const int* __restrict__ mask,
    __bf16* __restrict__ Qf, __bf16* __restrict__ Kft, int isK) {
  int wv = threadIdx.x >> 5;
  int lane = threadIdx.x & 31;
  int wid = blockIdx.x * 4 + wv;
  const int NT = NN / 16;
  if (wid >= BB * HH * NT) return;
  int nt = wid % NT;
  int bh = wid / NT;
  int h = bh % HH, b = bh / HH;
  int n0 = nt * 16;

  __shared__ __bf16 smem[4][16 * 256];
  __bf16* s = smem[wv];

  const float* Abase = qkv + ((size_t)b * NN + n0) * TD + (isK ? DD : 0) + h * DKK;
  const __bf16* Obase = omegab + (size_t)h * MM * DKK;

  v16bf a0 = load_a_f32(Abase, TD, 0,  lane);
  v16bf a1 = load_a_f32(Abase, TD, 32, lane);

  float ssq = 0.0f;
#pragma unroll
  for (int i = 0; i < 16; ++i) {
    float x0 = (float)a0[i]; ssq += x0 * x0;
    float x1 = (float)a1[i]; ssq += x1 * x1;
  }
  ssq += __shfl_xor(ssq, 16, 32);
  float sq = 0.5f * ssq;

  v8f acc[16];
#pragma unroll
  for (int t = 0; t < 16; ++t) {
    const __bf16* Ot = Obase + (size_t)t * 16 * DKK;
    v8f cfr = {};
    cfr = wmma_bf16(a0, load_b_bf16t(Ot, DKK, 0,  lane), cfr);
    cfr = wmma_bf16(a1, load_b_bf16t(Ot, DKK, 32, lane), cfr);
    acc[t] = cfr;
  }

  float mx[8];
#pragma unroll
  for (int v = 0; v < 8; ++v) {
    float m = -3.0e38f;
#pragma unroll
    for (int t = 0; t < 16; ++t) m = fmaxf(m, acc[t][v]);
    mx[v] = m;
  }
#pragma unroll
  for (int d = 1; d < 16; d <<= 1) {
#pragma unroll
    for (int v = 0; v < 8; ++v) mx[v] = fmaxf(mx[v], __shfl_xor(mx[v], d, 32));
  }

  const float inv_sqrt_M = 0.062499999881f;  // 1/sqrt(M + 1e-6)
  int half = lane >> 4, c0 = lane & 15;
  size_t bhoff = (size_t)bh;
#pragma unroll
  for (int v = 0; v < 8; ++v) {
    int nl = v + 8 * half;
    float sqv = __shfl(sq, nl, 32);
    float vm = 1.0f;
    if (isK) vm = mask[b * NN + n0 + nl] ? 0.0f : 1.0f;
#pragma unroll
    for (int t = 0; t < 16; ++t) {
      int m = t * 16 + c0;
      float e = __expf(acc[t][v] - mx[v] - sqv) * inv_sqrt_M * vm;
      if (isK) s[m * 16 + nl] = (__bf16)e;    // [m][n_local]
      else     s[nl * 256 + m] = (__bf16)e;   // [n_local][m]
    }
  }
  // Coalesced 16B global stores from LDS (same-wave producer/consumer).
  if (isK) {
#pragma unroll
    for (int i = 0; i < 8; ++i) {
      int m = i * 32 + lane;
      const __bf16* rp = s + m * 16;
      v8bf lo, hi;
#pragma unroll
      for (int jj = 0; jj < 8; ++jj) { lo[jj] = rp[jj]; hi[jj] = rp[8 + jj]; }
      __bf16* gp = Kft + (bhoff * MM + m) * NN + n0;
      *(v8bf*)(gp)     = lo;
      *(v8bf*)(gp + 8) = hi;
    }
  } else {
    int nl = lane >> 1, mh = lane & 1;
    const __bf16* rp = s + nl * 256 + mh * 128;
    __bf16* gp = Qf + (bhoff * NN + n0 + nl) * MM + mh * 128;
#pragma unroll
    for (int ii = 0; ii < 16; ++ii) {
      v8bf t;
#pragma unroll
      for (int jj = 0; jj < 8; ++jj) t[jj] = rp[ii * 8 + jj];
      *(v8bf*)(gp + ii * 8) = t;
    }
  }
}

// ---- Kernel 3: St[d][m] = (Kf^T V)^T in bf16, fused K_sum -----------------
__global__ void s_kernel(const __bf16* __restrict__ Kft, const __bf16* __restrict__ Vt,
                         __bf16* __restrict__ St, float* __restrict__ Ksum) {
  int wid  = (int)((blockIdx.x * blockDim.x + threadIdx.x) >> 5);
  int lane = threadIdx.x & 31;
  if (wid >= BB * HH * 16 * 4) return;
  int dt = wid & 3;
  int mt = (wid >> 2) & 15;
  int bh = wid >> 6;

  size_t bhoff = (size_t)bh;
  const __bf16* Abase = Kft + (bhoff * MM + mt * 16) * NN;        // rows m, stride N
  const __bf16* Bbase = Vt + (bhoff * DKK + dt * 16) * NN;        // rows d, stride N

  v8f acc = {};
  float ks = 0.0f;
  for (int k0 = 0; k0 < NN; k0 += 32) {
    v16bf a = load_a_bf16(Abase, NN, k0, lane);
    v16bf bb = load_b_bf16t(Bbase, NN, k0, lane);
    acc = wmma_bf16(a, bb, acc);
    if (dt == 0) {
#pragma unroll
      for (int i = 0; i < 16; ++i) ks += (float)a[i];
    }
  }
  // D tile: lane holds col d fixed, rows m_local = 8*half + v (consecutive!)
  int half = lane >> 4, c = lane & 15;
  __bf16* dst = St + (bhoff * DKK + dt * 16 + c) * MM + mt * 16 + 8 * half;
  v8bf t;
#pragma unroll
  for (int v = 0; v < 8; ++v) t[v] = (__bf16)acc[v];
  *(v8bf*)dst = t;
  if (dt == 0) {
    ks += __shfl_xor(ks, 16, 32);
    if (lane < 16) Ksum[bhoff * MM + mt * 16 + lane] = ks;
  }
}

// ---- Kernel 4: ctx = (Qf @ S) / (Qf . K_sum + eps), masked, bf16 out ------
__global__ void ctx_kernel(const __bf16* __restrict__ Qf, const __bf16* __restrict__ St,
                           const float* __restrict__ Ksum, const int* __restrict__ mask,
                           __bf16* __restrict__ ctxb) {
  int wid  = (int)((blockIdx.x * blockDim.x + threadIdx.x) >> 5);
  int lane = threadIdx.x & 31;
  const int NT = NN / 16;
  if (wid >= BB * HH * NT) return;
  int nt = wid % NT;
  int bh = wid / NT;
  int h = bh % HH, b = bh / HH;

  size_t bhoff = (size_t)bh;
  const __bf16* Abase = Qf + (bhoff * NN + nt * 16) * MM;   // rows n, stride M
  const __bf16* Sbase = St + bhoff * DKK * MM;              // rows d, stride M
  const float*  KS    = Ksum + bhoff * MM;

  v8f acc[4] = {};
  float dn = 0.0f;
  int half = lane >> 4;
  for (int k0 = 0; k0 < MM; k0 += 32) {
    v16bf a = load_a_bf16(Abase, MM, k0, lane);
#pragma unroll
    for (int g = 0; g < 2; ++g) {   // denom: frag elems 8g..8g+7 <-> 8 consec m's
      const float4* p = (const float4*)(KS + k0 + g * 16 + half * 8);
      float4 u = p[0], w = p[1];
      dn += (float)a[8*g+0]*u.x + (float)a[8*g+1]*u.y + (float)a[8*g+2]*u.z + (float)a[8*g+3]*u.w;
      dn += (float)a[8*g+4]*w.x + (float)a[8*g+5]*w.y + (float)a[8*g+6]*w.z + (float)a[8*g+7]*w.w;
    }
#pragma unroll
    for (int dt = 0; dt < 4; ++dt)
      acc[dt] = wmma_bf16(a, load_b_bf16t(Sbase + (size_t)dt * 16 * MM, MM, k0, lane), acc[dt]);
  }
  dn += __shfl_xor(dn, 16, 32);

  int c = lane & 15;
#pragma unroll
  for (int v = 0; v < 8; ++v) {
    int n = nt * 16 + v + 8 * half;
    float dnv = __shfl(dn, v + 8 * half, 32);
    float vm = mask[b * NN + n] ? 0.0f : 1.0f;
    float inv = vm / (dnv + EPSF);
#pragma unroll
    for (int dt = 0; dt < 4; ++dt)
      ctxb[((size_t)b * NN + n) * DD + h * DKK + dt * 16 + c] = (__bf16)(acc[dt][v] * inv);
  }
}

// ---- Kernel 5: out = ctx @ Wout^T + bout (16x64 per wave) -----------------
__global__ void out_kernel(const __bf16* __restrict__ ctxb, const __bf16* __restrict__ Wb,
                           const float* __restrict__ bias, float* __restrict__ out) {
  int wid  = (int)((blockIdx.x * blockDim.x + threadIdx.x) >> 5);
  int lane = threadIdx.x & 31;
  const int RT = BN / 16, CTG = DD / 64;  // 1024 x 16
  if (wid >= RT * CTG) return;
  int rt = wid % RT, ct0 = (wid / RT) * 4;

  const __bf16* Abase = ctxb + (size_t)rt * 16 * DD;
  v8f acc[4] = {};
  for (int k0 = 0; k0 < DD; k0 += 32) {
    v16bf a = load_a_bf16(Abase, DD, k0, lane);
    __builtin_prefetch(Abase + (size_t)(lane & 15) * DD + k0 + 128, 0, 1);
#pragma unroll
    for (int cc = 0; cc < 4; ++cc)
      acc[cc] = wmma_bf16(a, load_b_bf16t(Wb + (size_t)(ct0 + cc) * 16 * DD, DD, k0, lane), acc[cc]);
  }
  int c = lane & 15, half = lane >> 4;
#pragma unroll
  for (int v = 0; v < 8; ++v) {
    int rowg = rt * 16 + v + 8 * half;
#pragma unroll
    for (int cc = 0; cc < 4; ++cc) {
      int j = (ct0 + cc) * 16 + c;
      out[(size_t)rowg * DD + j] = acc[cc][v] + bias[j];
    }
  }
}

// ---- Launch ---------------------------------------------------------------
extern "C" void kernel_launch(void* const* d_in, const int* in_sizes, int n_in,
                              void* d_out, int out_size, void* d_ws, size_t ws_size,
                              hipStream_t stream) {
  const float* x     = (const float*)d_in[0];
  const float* Wqkv  = (const float*)d_in[1];
  const float* bqkv  = (const float*)d_in[2];
  const float* Wout  = (const float*)d_in[3];
  const float* bout  = (const float*)d_in[4];
  const float* omega = (const float*)d_in[5];
  const int*   mask  = (const int*)d_in[6];
  float* out = (float*)d_out;

  char* ws = (char*)d_ws;
  float*  qkv    = (float*)(ws);                    // (BN,3D) f32     192 MB
  __bf16* xb     = (__bf16*)(ws + 201326592ull);    // (BN,D) bf16      32 MB
  __bf16* Qf     = (__bf16*)(ws + 234881024ull);    // (B,H,N,M) bf16   32 MB
  __bf16* Kft    = (__bf16*)(ws + 268435456ull);    // (B,H,M,N) bf16   32 MB
  __bf16* Vt     = (__bf16*)(ws + 301989888ull);    // (B,H,dk,N) bf16  32 MB
  __bf16* Stb    = (__bf16*)(ws + 335544320ull);    // (B,H,dk,M) bf16   2 MB
  float*  Ksum   = (float*)(ws + 337641472ull);     // (B,H,M) f32      64 KB
  __bf16* ctxb   = (__bf16*)(ws + 337707008ull);    // (BN,D) bf16      32 MB
  __bf16* Wqkvb  = (__bf16*)(ws + 371261440ull);    // (3D,D) bf16       6 MB
  __bf16* Woutb  = (__bf16*)(ws + 377552896ull);    // (D,D) bf16        2 MB
  __bf16* omegab = (__bf16*)(ws + 379650048ull);    // (H,M,dk) bf16   0.5 MB

  dim3 blk(256);
  // Pre-convert operands to bf16 (one pass each; removes cvt from GEMM loops)
  cvt_kernel<<<8192, blk, 0, stream>>>(x,     xb,     BN * DD / 8);
  cvt_kernel<<<1536, blk, 0, stream>>>(Wqkv,  Wqkvb,  TD * DD / 8);
  cvt_kernel<<<512,  blk, 0, stream>>>(Wout,  Woutb,  DD * DD / 8);
  cvt_kernel<<<128,  blk, 0, stream>>>(omega, omegab, HH * MM * DKK / 8);

  qkv_kernel<<<6144, blk, 0, stream>>>(xb, Wqkvb, bqkv, mask, qkv);
  vt_kernel<<<8192, blk, 0, stream>>>(qkv, Vt);
  phi_kernel<<<4096, dim3(128), 0, stream>>>(qkv, omegab, mask, Qf, Kft, 0);
  phi_kernel<<<4096, dim3(128), 0, stream>>>(qkv, omegab, mask, Qf, Kft, 1);
  s_kernel<<<512, blk, 0, stream>>>(Kft, Vt, Stb, Ksum);
  ctx_kernel<<<2048, blk, 0, stream>>>(Qf, Stb, Ksum, mask, ctxb);
  out_kernel<<<2048, blk, 0, stream>>>(ctxb, Woutb, bout, out);
}
